// CapsuleLayer_32444182954357
// MI455X (gfx1250) — matrix operand order repaired and checked by reading, not compiled, not run on hardware
//
#include <hip/hip_runtime.h>
#include <math.h>

typedef unsigned int  uint_t;
typedef unsigned short ushort_t;

typedef __attribute__((ext_vector_type(16))) __bf16 v16bf;
typedef __attribute__((ext_vector_type(8)))  float  v8f;

union FragBF { uint4 q[2]; v16bf v; };

#define IU 1024   // in_units
#define CH 128    // in_channels (K)
#define DD 64     // dim_capsules
#define JC 32     // num_capsules
#define BA 32     // batch (M)

// ---------- helpers ----------
__device__ __forceinline__ ushort_t f2bf(float f) {
  uint_t u = __float_as_uint(f);
  u += 0x7fffu + ((u >> 16) & 1u);            // round-to-nearest-even
  return (ushort_t)(u >> 16);
}
// packed fp32x2 -> bf16x2 in one VALU op
__device__ __forceinline__ uint_t pack2bf(float lo, float hi) {
  uint_t r;
  asm("v_cvt_pk_bf16_f32 %0, %1, %2" : "=v"(r) : "v"(lo), "v"(hi));
  return r;
}
__device__ __forceinline__ float bflo(uint_t u) { return __uint_as_float(u << 16); }
__device__ __forceinline__ float bfhi(uint_t u) { return __uint_as_float(u & 0xffff0000u); }

__device__ __forceinline__ float wsum(float v) {
#pragma unroll
  for (int m = 16; m >= 1; m >>= 1) v += __shfl_xor(v, m, 32);
  return v;
}
__device__ __forceinline__ float wmax(float v) {
#pragma unroll
  for (int m = 16; m >= 1; m >>= 1) v = fmaxf(v, __shfl_xor(v, m, 32));
  return v;
}

// ---------- kernel 1: u_hat via bf16 WMMA ----------
// grid (IU, 8): block handles i = blockIdx.x, 4 capsules j0..j0+3 (N = 4*64 = 256)
// GEMM: M=32 (batch), N=256 (j,d), K=128 (c). 8 wave32's, each wave: 1 M-tile x 4 N-tiles.
__global__ __launch_bounds__(256)
void gemm_uhat(const float* __restrict__ X, const float* __restrict__ W,
               ushort_t* __restrict__ U) {
  const int i  = blockIdx.x;
  const int j0 = blockIdx.y * 4;
  const int tid = threadIdx.x;

  __shared__ __align__(16) ushort_t Ab[32][136];   // [m][k] bf16, padded
  __shared__ __align__(16) ushort_t Bb[256][136];  // [n][k] bf16, padded

  const int c = (tid & 31) * 4;  // fixed 4-channel column per thread

  // stage X row-block (32x128 fp32 -> bf16)
  {
    const int b0 = tid >> 5;
#pragma unroll
    for (int r = 0; r < 4; ++r) {
      const int b = b0 + r * 8;
      const float4 f = *(const float4*)(X + ((size_t)b * IU + i) * CH + c);
      uint2 p; p.x = pack2bf(f.x, f.y); p.y = pack2bf(f.z, f.w);
      *(uint2*)&Ab[b][c] = p;
    }
  }
  // stage W slice (256x128 fp32 -> bf16), [n][k] layout (n=(j,d), k=c)
  {
    const int n0 = tid >> 5;
#pragma unroll 4
    for (int r = 0; r < 32; ++r) {
      const int n = n0 + r * 8;
      const int j = j0 + (n >> 6);
      const int d = n & 63;
      const float4 f = *(const float4*)(W + (((size_t)j * IU + i) * DD + d) * CH + c);
      uint2 p; p.x = pack2bf(f.x, f.y); p.y = pack2bf(f.z, f.w);
      *(uint2*)&Bb[n][c] = p;
    }
  }
  __syncthreads();

  const int wave  = tid >> 5;
  const int lane  = tid & 31;
  const int mt    = wave & 1;           // which 16-row M tile
  const int nt0   = (wave >> 1) * 4;    // first of 4 N tiles
  const int row   = (lane & 15) + mt * 16;
  const int khalf = lane >> 4;
  const int ncol  = lane & 15;

  v8f acc[4];
#pragma unroll
  for (int q = 0; q < 4; ++q) {
#pragma unroll
    for (int e = 0; e < 8; ++e) acc[q][e] = 0.0f;
  }

#pragma unroll
  for (int ks = 0; ks < 4; ++ks) {            // K steps of 32
    const int kb = ks * 32;
    FragBF fa;
    // A frag (16x32 bf16): VGPR0-3 -> K[kb+khalf*8 .. +7], VGPR4-7 -> K[kb+16+khalf*8 .. +7]
    fa.q[0] = *(const uint4*)&Ab[row][kb + khalf * 8];
    fa.q[1] = *(const uint4*)&Ab[row][kb + 16 + khalf * 8];
#pragma unroll
    for (int q = 0; q < 4; ++q) {
      const int n = (nt0 + q) * 16 + ncol;
      FragBF fb;
      // B frag (32x16 bf16): VGPR0-3 -> K[kb+khalf*8 .. +7], VGPR4-7 -> K[kb+16+khalf*8 .. +7]
      fb.q[0] = *(const uint4*)&Bb[n][kb + khalf * 8];
      fb.q[1] = *(const uint4*)&Bb[n][kb + 16 + khalf * 8];
      acc[q] = __builtin_amdgcn_wmma_f32_16x16x32_bf16(
          false, fa.v, false, fb.v, (short)0, acc[q], false, false);
    }
  }

  // write u_hat[b][j][i][d] as bf16. C/D layout: VGPR r -> M = r + 8*khalf (+16*mt)
  const size_t bstride = (size_t)JC * IU * DD;  // stride of batch index
  const int brow = mt * 16 + khalf * 8;
#pragma unroll
  for (int q = 0; q < 4; ++q) {
    const int n = (nt0 + q) * 16 + ncol;
    const int j = j0 + (n >> 6);
    const int d = n & 63;
    const size_t base = ((size_t)j * IU + i) * DD + d;
#pragma unroll
    for (int r = 0; r < 8; ++r) {
      U[(size_t)(brow + r) * bstride + base] = f2bf(acc[q][r]);
    }
  }
}

// ---------- kernel 2: fused routing pass ----------
// grid (64, 32): b = blockIdx.y, i-chunk of 16 = blockIdx.x.
// One HBM read of the u_hat tile serves agreement, softmax, and c*u accumulation.
// The bf16 tile is staged with CDNA5 async global->LDS copies (no VGPR round-trip).
__global__ __launch_bounds__(256)
void route_pass(const ushort_t* __restrict__ U, float* __restrict__ bij,
                const float* __restrict__ vprev, float* __restrict__ s, int t) {
  const int b  = blockIdx.y;
  const int i0 = blockIdx.x * 16;
  const int tid = threadIdx.x;

  __shared__ __align__(16) ushort_t Us[32 * 16 * 64];  // [j][ii][d] bf16 (64 KB)
  __shared__ float Vs[JC * DD];                        // v_prev[b] slice
  __shared__ float Cs[16][32];                         // softmax coeffs [ii][j]
  __shared__ float Bn[16][32];                         // updated logits [ii][j]

  if (t > 0) {
    for (int r = tid; r < JC * DD; r += 256) Vs[r] = vprev[(size_t)b * JC * DD + r];
  }
  // stage u_hat tile: per (j,ii) 64 bf16 = 8 x 16B async transfers (ASYNCcnt-tracked)
  for (int r = tid; r < 4096; r += 256) {
    const int jii = r >> 3;            // j*16 + ii
    const int q   = r & 7;
    const int j   = jii >> 4;
    const int ii  = jii & 15;
    const ushort_t* g =
        U + (((size_t)b * JC + j) * IU + (i0 + ii)) * DD + q * 8;
    // flat->LDS aperture keeps the workgroup-relative LDS offset in addr[31:0]
    const uint_t lds = (uint_t)(unsigned long long)(const void*)&Us[(j * 16 + ii) * 64 + q * 8];
    asm volatile("global_load_async_to_lds_b128 %0, %1, off"
                 :: "v"(lds), "v"((unsigned long long)g)
                 : "memory");
  }
  asm volatile("s_wait_asynccnt 0x0" ::: "memory");
  __syncthreads();

  const int wave = tid >> 5;
  const int lane = tid & 31;

  if (t > 0) {
    // Phase A: agreement + logit update, one (ii,j) pair per wave iteration
    for (int p = wave; p < 512; p += 8) {
      const int ii = p >> 5;
      const int j  = p & 31;
      const uint_t u2 = *(const uint_t*)&Us[(j * 16 + ii) * 64 + 2 * lane];
      float a = bflo(u2) * Vs[j * DD + 2 * lane] + bfhi(u2) * Vs[j * DD + 2 * lane + 1];
      a = wsum(a);
      if (lane == 0) {
        const size_t bi = ((size_t)b * JC + j) * IU + (i0 + ii);
        const float bn = bij[bi] + a;
        bij[bi]   = bn;
        Bn[ii][j] = bn;
      }
    }
    __syncthreads();
    // Phase B: softmax over j per (b,i)
    for (int ii = wave; ii < 16; ii += 8) {
      const float x = Bn[ii][lane];
      const float m = wmax(x);
      const float e = __expf(x - m);
      const float z = wsum(e);
      Cs[ii][lane]  = e / z;
    }
    __syncthreads();
  } else {
    for (int r = tid; r < 512; r += 256) ((float*)Cs)[r] = 1.0f / 32.0f;
    __syncthreads();
  }

  // Phase C: s[j][d] partial = sum_ii c[ii][j] * u[j][ii][d]
#pragma unroll
  for (int qq = 0; qq < 4; ++qq) {
    const int j = wave * 4 + qq;
    float a0 = 0.f, a1 = 0.f;
#pragma unroll 4
    for (int ii = 0; ii < 16; ++ii) {
      const uint_t u2 = *(const uint_t*)&Us[(j * 16 + ii) * 64 + 2 * lane];
      const float cc = Cs[ii][j];
      a0 += cc * bflo(u2);
      a1 += cc * bfhi(u2);
    }
    float* sp = s + ((size_t)b * JC + j) * DD + 2 * lane;
    atomicAdd(sp, a0);
    atomicAdd(sp + 1, a1);
  }
}

// ---------- kernel 3: squash ----------
// one wave per (b,j) vector of 64
__global__ __launch_bounds__(256)
void squash_kernel(const float* __restrict__ s, float* __restrict__ v) {
  const int tid  = threadIdx.x;
  const int wave = tid >> 5;
  const int lane = tid & 31;
  const int bj   = blockIdx.x * 8 + wave;  // 0..1023
  const float* sp = s + (size_t)bj * DD + 2 * lane;
  const float x0 = sp[0], x1 = sp[1];
  const float tot = wsum(x0 * x0 + x1 * x1);   // ||s||^2
  const float nrm = sqrtf(tot);
  const float scale = nrm / (1.0f + tot);
  float* vp = v + (size_t)bj * DD + 2 * lane;
  vp[0] = x0 * scale;
  vp[1] = x1 * scale;
}

// ---------- kernel 4: zero scratch (b_ij + s0..s2), re-run every launch ----------
__global__ void zero_kernel(float* __restrict__ p, int n) {
  const int idx = blockIdx.x * blockDim.x + threadIdx.x;
  if (idx < n) p[idx] = 0.0f;
}

extern "C" void kernel_launch(void* const* d_in, const int* in_sizes, int n_in,
                              void* d_out, int out_size, void* d_ws, size_t ws_size,
                              hipStream_t stream) {
  const float* x = (const float*)d_in[0];  // (B, I, C) fp32
  const float* W = (const float*)d_in[1];  // (J, I, D, C) fp32
  float* out = (float*)d_out;              // (B, J, D) fp32

  char* ws = (char*)d_ws;
  // workspace layout
  ushort_t* U  = (ushort_t*)ws;                         // u_hat bf16: 134,217,728 B
  float* bij   = (float*)(ws + 134217728ull);           // 1,048,576 f32
  float* s0    = (float*)(ws + 138412032ull);           // 65,536 f32 each
  float* s1    = s0 + 65536;
  float* s2    = s1 + 65536;
  float* v0    = s2 + 65536;
  float* v1    = v0 + 65536;

  const int nzero = 1048576 + 3 * 65536;  // b_ij + s0..s2
  zero_kernel<<<(nzero + 255) / 256, 256, 0, stream>>>(bij, nzero);

  gemm_uhat<<<dim3(IU, 8), 256, 0, stream>>>(x, W, U);

  route_pass<<<dim3(64, 32), 256, 0, stream>>>(U, bij, nullptr, s0, 0);
  squash_kernel<<<128, 256, 0, stream>>>(s0, v0);
  route_pass<<<dim3(64, 32), 256, 0, stream>>>(U, bij, v0, s1, 1);
  squash_kernel<<<128, 256, 0, stream>>>(s1, v1);
  route_pass<<<dim3(64, 32), 256, 0, stream>>>(U, bij, v1, s2, 2);
  squash_kernel<<<128, 256, 0, stream>>>(s2, out);
}
